// TAGUNetBlock_15839839388406
// MI455X (gfx1250) — matrix operand order repaired and checked by reading, not compiled, not run on hardware
//
#include <hip/hip_runtime.h>
#include <hip/hip_bf16.h>
#include <stdint.h>

// ---------------- sizes from reference ----------------
#define N0_NODES 50000
#define N1_NODES 12500
#define E0_EDGES 1600000
#define E1_EDGES 400000
#define BN_EPS   1e-5f

typedef __attribute__((ext_vector_type(16))) _Float16 v16h;
typedef __attribute__((ext_vector_type(8)))  float    v8f;
typedef __attribute__((ext_vector_type(8)))  unsigned int v8u;

union H2U { _Float16 h[2]; unsigned int u; };

// ---------------- WMMA helpers (gfx1250, wave32) ----------------
__device__ __forceinline__ v8f wmma_f16(v16h a, v16h b, v8f c) {
  // D = A(16x32 f16) * B(32x16 f16) + C(16x16 f32)
  return __builtin_amdgcn_wmma_f32_16x16x32_f16(false, a, false, b, (short)0, c,
                                                false, false);
}

// A fragment 16x32 f16 from row-major LDS [16][ldk] (ldk in f16 elements, even).
// ISA layout: lanes 0-15 -> rows 0-15, K=0..7 & 16..23; lanes 16-31 -> K=8..15 & 24..31.
__device__ __forceinline__ v16h load_a_frag(const unsigned int* lds32, int lane,
                                            int ldk, int kt) {
  int row  = lane & 15;
  int kb   = ((lane >> 4) << 3) + (kt << 5);
  int base = (row * ldk + kb) >> 1;
  v8u a;
#pragma unroll
  for (int v = 0; v < 4; ++v) a[v] = lds32[base + v];
#pragma unroll
  for (int v = 4; v < 8; ++v) a[v] = lds32[base + v + 4];
  return __builtin_bit_cast(v16h, a);
}

// B fragment from pre-packed (fragment-major) weight buffer: 8 contiguous dwords/lane.
__device__ __forceinline__ v16h load_b_frag(const unsigned int* __restrict__ pw,
                                            int lane, int ntiles, int kt, int nt) {
  const unsigned int* f = pw + (((size_t)(kt * ntiles + nt) * 32 + lane) << 3);
  v8u b;
#pragma unroll
  for (int v = 0; v < 8; ++v) b[v] = f[v];
  return __builtin_bit_cast(v16h, b);
}

// ---------------- weight packing: f32 [K,N] row-major -> fragment-major f16 ----------------
// layout: dst[((kt*ntiles+nt)*32 + lane)*8 + v] = pack(W[kt*32+(lane>>4)*8+2v+(v>=4?8:0)][nt*16+(lane&15)],
//                                                      W[...+1][...])
__global__ void pack_weight_kernel(const float* __restrict__ W, int K, int N,
                                   unsigned int* __restrict__ dst) {
  int total = (K * N) >> 1;
  int idx = blockIdx.x * blockDim.x + threadIdx.x;
  if (idx >= total) return;
  int ntiles = N >> 4;
  int v    = idx & 7;
  int lane = (idx >> 3) & 31;
  int frag = idx >> 8;
  int kt = frag / ntiles;
  int nt = frag - kt * ntiles;
  int col = (nt << 4) + (lane & 15);
  int k0  = (kt << 5) + ((lane >> 4) << 3) + (v << 1) + ((v >> 2) << 3);
  H2U p;
  p.h[0] = (_Float16)W[(size_t)k0 * N + col];
  p.h[1] = (_Float16)W[(size_t)(k0 + 1) * N + col];
  dst[idx] = p.u;
}

// ---------------- fused EdgeConv (3-layer MLP + relu + atomic max-agg) ----------------
// per wave: 16 edges. input per edge = [x_dst, x_src - x_dst] (2*DIN), layers:
// 2*DIN ->64 relu ->64 relu ->64, then relu + atomicMax into agg[dst].
template <int DIN>
__global__ __launch_bounds__(128) void edge_conv_kernel(
    const float* __restrict__ Xin, const int* __restrict__ edges, long E,
    const unsigned int* __restrict__ pw0, const float* __restrict__ b0,
    const unsigned int* __restrict__ pw1, const float* __restrict__ b1,
    const unsigned int* __restrict__ pw2, const float* __restrict__ b2,
    float* __restrict__ agg) {
  constexpr int DCAT = 2 * DIN;
  constexpr int KT0  = DCAT / 32;
  __shared__ _Float16 uSh[4][16][DCAT];
  __shared__ _Float16 hSh[4][2][16][64];
  int w    = threadIdx.x >> 5;
  int lane = threadIdx.x & 31;
  long blockBase = (long)blockIdx.x * 64 + (long)w * 16;

  // ---- gather edge features into LDS (f16) ----
  for (int e = 0; e < 16; ++e) {
    long ed = blockBase + e;
    if (ed < E) {
      int s = edges[ed];
      int d = edges[E + ed];
      const float* xs = Xin + (size_t)s * DIN;
      const float* xd = Xin + (size_t)d * DIN;
      for (int c = lane; c < DIN; c += 32) {
        float xi = xd[c];
        float xj = xs[c];
        uSh[w][e][c]       = (_Float16)xi;
        uSh[w][e][DIN + c] = (_Float16)(xj - xi);
      }
    } else {
      for (int c = lane; c < DCAT; c += 32) uSh[w][e][c] = (_Float16)0.0f;
    }
  }
  __syncthreads();

  int col = lane & 15;
  int rb  = (lane >> 4) << 3;

  // ---- layer 0: DCAT -> 64, relu ----
  {
    v8f acc[4];
#pragma unroll
    for (int nt = 0; nt < 4; ++nt)
#pragma unroll
      for (int i = 0; i < 8; ++i) acc[nt][i] = 0.0f;
    const unsigned int* u32 = (const unsigned int*)&uSh[w][0][0];
#pragma unroll
    for (int kt = 0; kt < KT0; ++kt) {
      v16h a = load_a_frag(u32, lane, DCAT, kt);
#pragma unroll
      for (int nt = 0; nt < 4; ++nt) {
        v16h b = load_b_frag(pw0, lane, 4, kt, nt);
        acc[nt] = wmma_f16(a, b, acc[nt]);
      }
    }
#pragma unroll
    for (int nt = 0; nt < 4; ++nt) {
      float bias = b0[nt * 16 + col];
#pragma unroll
      for (int v = 0; v < 8; ++v)
        hSh[w][0][rb + v][nt * 16 + col] = (_Float16)fmaxf(acc[nt][v] + bias, 0.0f);
    }
  }
  __syncthreads();

  // ---- layer 1: 64 -> 64, relu ----
  {
    v8f acc[4];
#pragma unroll
    for (int nt = 0; nt < 4; ++nt)
#pragma unroll
      for (int i = 0; i < 8; ++i) acc[nt][i] = 0.0f;
    const unsigned int* u32 = (const unsigned int*)&hSh[w][0][0][0];
#pragma unroll
    for (int kt = 0; kt < 2; ++kt) {
      v16h a = load_a_frag(u32, lane, 64, kt);
#pragma unroll
      for (int nt = 0; nt < 4; ++nt) {
        v16h b = load_b_frag(pw1, lane, 4, kt, nt);
        acc[nt] = wmma_f16(a, b, acc[nt]);
      }
    }
#pragma unroll
    for (int nt = 0; nt < 4; ++nt) {
      float bias = b1[nt * 16 + col];
#pragma unroll
      for (int v = 0; v < 8; ++v)
        hSh[w][1][rb + v][nt * 16 + col] = (_Float16)fmaxf(acc[nt][v] + bias, 0.0f);
    }
  }
  __syncthreads();

  // ---- layer 2: 64 -> 64, then relu + atomic max scatter ----
  {
    v8f acc[4];
#pragma unroll
    for (int nt = 0; nt < 4; ++nt)
#pragma unroll
      for (int i = 0; i < 8; ++i) acc[nt][i] = 0.0f;
    const unsigned int* u32 = (const unsigned int*)&hSh[w][1][0][0];
#pragma unroll
    for (int kt = 0; kt < 2; ++kt) {
      v16h a = load_a_frag(u32, lane, 64, kt);
#pragma unroll
      for (int nt = 0; nt < 4; ++nt) {
        v16h b = load_b_frag(pw2, lane, 4, kt, nt);
        acc[nt] = wmma_f16(a, b, acc[nt]);
      }
    }
#pragma unroll
    for (int nt = 0; nt < 4; ++nt) {
      int ch = nt * 16 + col;
      float bias = b2[ch];
#pragma unroll
      for (int v = 0; v < 8; ++v) {
        long ed = blockBase + rb + v;
        if (ed < E) {
          // relu folded into max-agg: agg initialized to 0; nonneg float bits
          // compare correctly as int -> global_atomic_max_i32.
          float val = fmaxf(acc[nt][v] + bias, 0.0f);
          int d = edges[E + ed];
          atomicMax(reinterpret_cast<int*>(agg) + ((size_t)d * 64 + ch),
                    __float_as_int(val));
        }
      }
    }
  }
}

// ---------------- fused output MLP: 64 ->128 relu ->128 relu ->64 ----------------
__global__ __launch_bounds__(128) void mlp_out_kernel(
    const float* __restrict__ Xin, int nrows,
    const unsigned int* __restrict__ pw0, const float* __restrict__ b0,
    const unsigned int* __restrict__ pw1, const float* __restrict__ b1,
    const unsigned int* __restrict__ pw2, const float* __restrict__ b2,
    float* __restrict__ out) {
  __shared__ _Float16 inSh[4][16][64];
  __shared__ _Float16 h1Sh[4][16][128];
  __shared__ _Float16 h2Sh[4][16][128];
  int w    = threadIdx.x >> 5;
  int lane = threadIdx.x & 31;
  long rowBase = (long)blockIdx.x * 64 + (long)w * 16;

  for (int r = 0; r < 16; ++r) {
    long row = rowBase + r;
    if (row < nrows) {
      for (int c = lane; c < 64; c += 32) inSh[w][r][c] = (_Float16)Xin[row * 64 + c];
    } else {
      for (int c = lane; c < 64; c += 32) inSh[w][r][c] = (_Float16)0.0f;
    }
  }
  __syncthreads();

  int col = lane & 15;
  int rb  = (lane >> 4) << 3;

  // layer 1: 64 -> 128, relu
  {
    v8f acc[8];
#pragma unroll
    for (int nt = 0; nt < 8; ++nt)
#pragma unroll
      for (int i = 0; i < 8; ++i) acc[nt][i] = 0.0f;
    const unsigned int* u32 = (const unsigned int*)&inSh[w][0][0];
#pragma unroll
    for (int kt = 0; kt < 2; ++kt) {
      v16h a = load_a_frag(u32, lane, 64, kt);
#pragma unroll
      for (int nt = 0; nt < 8; ++nt) {
        v16h b = load_b_frag(pw0, lane, 8, kt, nt);
        acc[nt] = wmma_f16(a, b, acc[nt]);
      }
    }
#pragma unroll
    for (int nt = 0; nt < 8; ++nt) {
      float bias = b0[nt * 16 + col];
#pragma unroll
      for (int v = 0; v < 8; ++v)
        h1Sh[w][rb + v][nt * 16 + col] = (_Float16)fmaxf(acc[nt][v] + bias, 0.0f);
    }
  }
  __syncthreads();

  // layer 2: 128 -> 128, relu
  {
    v8f acc[8];
#pragma unroll
    for (int nt = 0; nt < 8; ++nt)
#pragma unroll
      for (int i = 0; i < 8; ++i) acc[nt][i] = 0.0f;
    const unsigned int* u32 = (const unsigned int*)&h1Sh[w][0][0];
#pragma unroll
    for (int kt = 0; kt < 4; ++kt) {
      v16h a = load_a_frag(u32, lane, 128, kt);
#pragma unroll
      for (int nt = 0; nt < 8; ++nt) {
        v16h b = load_b_frag(pw1, lane, 8, kt, nt);
        acc[nt] = wmma_f16(a, b, acc[nt]);
      }
    }
#pragma unroll
    for (int nt = 0; nt < 8; ++nt) {
      float bias = b1[nt * 16 + col];
#pragma unroll
      for (int v = 0; v < 8; ++v)
        h2Sh[w][rb + v][nt * 16 + col] = (_Float16)fmaxf(acc[nt][v] + bias, 0.0f);
    }
  }
  __syncthreads();

  // layer 3: 128 -> 64, no relu, store
  {
    v8f acc[4];
#pragma unroll
    for (int nt = 0; nt < 4; ++nt)
#pragma unroll
      for (int i = 0; i < 8; ++i) acc[nt][i] = 0.0f;
    const unsigned int* u32 = (const unsigned int*)&h2Sh[w][0][0];
#pragma unroll
    for (int kt = 0; kt < 4; ++kt) {
      v16h a = load_a_frag(u32, lane, 128, kt);
#pragma unroll
      for (int nt = 0; nt < 4; ++nt) {
        v16h b = load_b_frag(pw2, lane, 4, kt, nt);
        acc[nt] = wmma_f16(a, b, acc[nt]);
      }
    }
#pragma unroll
    for (int nt = 0; nt < 4; ++nt) {
      int ch = nt * 16 + col;
      float bias = b2[ch];
#pragma unroll
      for (int v = 0; v < 8; ++v) {
        long row = rowBase + rb + v;
        if (row < nrows) out[(size_t)row * 64 + ch] = acc[nt][v] + bias;
      }
    }
  }
}

// ---------------- utility kernels ----------------
__global__ void zero_kernel(float* __restrict__ p, long n) {
  long i = (long)blockIdx.x * blockDim.x + threadIdx.x;
  if (i < n) p[i] = 0.0f;
}

__global__ __launch_bounds__(256) void bn_stats_kernel(const float* __restrict__ x,
                                                       int nrows,
                                                       float* __restrict__ stats) {
  __shared__ float sS[4][64];
  __shared__ float sQ[4][64];
  int ch  = threadIdx.x & 63;
  int grp = threadIdx.x >> 6;
  long r0 = (long)blockIdx.x * 2048;
  long r1 = r0 + 2048; if (r1 > nrows) r1 = nrows;
  float s = 0.0f, q = 0.0f;
  for (long r = r0 + grp; r < r1; r += 4) {
    float v = x[r * 64 + ch];
    s += v; q += v * v;
  }
  sS[grp][ch] = s; sQ[grp][ch] = q;
  __syncthreads();
  if (grp == 0) {
    s = sS[0][ch] + sS[1][ch] + sS[2][ch] + sS[3][ch];
    q = sQ[0][ch] + sQ[1][ch] + sQ[2][ch] + sQ[3][ch];
    atomicAdd(&stats[ch], s);
    atomicAdd(&stats[64 + ch], q);
  }
}

__global__ void bn_finalize_kernel(const float* __restrict__ stats,
                                   const float* __restrict__ gamma,
                                   const float* __restrict__ beta, float n,
                                   float* __restrict__ ss) {
  int ch = threadIdx.x;
  if (ch < 64) {
    float mu  = stats[ch] / n;
    float var = stats[64 + ch] / n - mu * mu;
    float sc  = gamma[ch] * rsqrtf(var + BN_EPS);
    ss[ch]      = sc;
    ss[64 + ch] = beta[ch] - mu * sc;
  }
}

__global__ void bn_apply_kernel(const float* __restrict__ x,
                                const float* __restrict__ ss,
                                float* __restrict__ y, long n) {
  long i = (long)blockIdx.x * blockDim.x + threadIdx.x;
  if (i < n) {
    int ch = (int)(i & 63);
    y[i] = x[i] * ss[ch] + ss[64 + ch];
  }
}

__global__ void pool_scatter_kernel(const float* __restrict__ x,
                                    const int* __restrict__ cluster,
                                    float* __restrict__ p, float* __restrict__ cnt,
                                    int nrows) {
  long i = (long)blockIdx.x * blockDim.x + threadIdx.x;
  if (i >= (long)nrows * 64) return;
  int node = (int)(i >> 6);
  int ch   = (int)(i & 63);
  int c = cluster[node];
  atomicAdd(&p[(size_t)c * 64 + ch], x[i]);
  if (ch == 0) atomicAdd(&cnt[c], 1.0f);
}

__global__ void pool_div_kernel(float* __restrict__ p, const float* __restrict__ cnt,
                                long n) {
  long i = (long)blockIdx.x * blockDim.x + threadIdx.x;
  if (i < n) p[i] /= fmaxf(cnt[i >> 6], 1.0f);
}

__global__ void cat_xc_kernel(const float* __restrict__ x1, float* __restrict__ xcin,
                              long n) {
  long i = (long)blockIdx.x * blockDim.x + threadIdx.x;
  if (i < n) {
    long node = i >> 7;
    int  c    = (int)(i & 127);
    xcin[i] = x1[node * 64 + (c & 63)];
  }
}

__global__ void cat_xf_kernel(const float* __restrict__ x0,
                              const float* __restrict__ xc,
                              const int* __restrict__ cluster,
                              float* __restrict__ xfin, long n) {
  long i = (long)blockIdx.x * blockDim.x + threadIdx.x;
  if (i < n) {
    long node = i >> 7;
    int  c    = (int)(i & 127);
    xfin[i] = (c < 64) ? x0[node * 64 + c]
                       : xc[(size_t)cluster[node] * 64 + (c - 64)];
  }
}

// ---------------- host orchestration ----------------
extern "C" void kernel_launch(void* const* d_in, const int* in_sizes, int n_in,
                              void* d_out, int out_size, void* d_ws, size_t ws_size,
                              hipStream_t stream) {
  (void)in_sizes; (void)n_in; (void)out_size; (void)ws_size;

  // ---- input pointer map (setup_inputs dict order) ----
  const float* X        = (const float*)d_in[0];
  const int*   edges0   = (const int*)d_in[1];
  const int*   edges1   = (const int*)d_in[2];
  const int*   cluster0 = (const int*)d_in[3];
  // Lconv0: 4..9, Lconv1: 10..15, Rconv0: 16..21, Rconv1: 22..27
  // Lnorm0: 28,29  Lnorm1: 30,31  Rnorm0: 32,33  Rnorm1: 34,35
  // mlp_out: 36..41
  const float* Lc0W[3] = {(const float*)d_in[4], (const float*)d_in[6], (const float*)d_in[8]};
  const float* Lc0b[3] = {(const float*)d_in[5], (const float*)d_in[7], (const float*)d_in[9]};
  const float* Lc1W[3] = {(const float*)d_in[10], (const float*)d_in[12], (const float*)d_in[14]};
  const float* Lc1b[3] = {(const float*)d_in[11], (const float*)d_in[13], (const float*)d_in[15]};
  const float* Rc0W[3] = {(const float*)d_in[16], (const float*)d_in[18], (const float*)d_in[20]};
  const float* Rc0b[3] = {(const float*)d_in[17], (const float*)d_in[19], (const float*)d_in[21]};
  const float* Rc1W[3] = {(const float*)d_in[22], (const float*)d_in[24], (const float*)d_in[26]};
  const float* Rc1b[3] = {(const float*)d_in[23], (const float*)d_in[25], (const float*)d_in[27]};
  const float* Lnorm0g = (const float*)d_in[28]; const float* Lnorm0b = (const float*)d_in[29];
  const float* Lnorm1g = (const float*)d_in[30]; const float* Lnorm1b = (const float*)d_in[31];
  const float* Rnorm0g = (const float*)d_in[32]; const float* Rnorm0b = (const float*)d_in[33];
  const float* Rnorm1g = (const float*)d_in[34]; const float* Rnorm1b = (const float*)d_in[35];
  const float* MW[3] = {(const float*)d_in[36], (const float*)d_in[38], (const float*)d_in[40]};
  const float* Mb[3] = {(const float*)d_in[37], (const float*)d_in[39], (const float*)d_in[41]};
  float* out = (float*)d_out;

  // ---- workspace carve-up ----
  char* wsp = (char*)d_ws;
  auto alloc = [&](size_t bytes) {
    char* p = wsp;
    wsp += (bytes + 255) & ~(size_t)255;
    return p;
  };
  unsigned int* pLc0[3]; unsigned int* pLc1[3]; unsigned int* pRc0[3]; unsigned int* pRc1[3];
  unsigned int* pM[3];
  pLc0[0] = (unsigned int*)alloc(128 * 64 * 2); pLc0[1] = (unsigned int*)alloc(64 * 64 * 2); pLc0[2] = (unsigned int*)alloc(64 * 64 * 2);
  pLc1[0] = (unsigned int*)alloc(128 * 64 * 2); pLc1[1] = (unsigned int*)alloc(64 * 64 * 2); pLc1[2] = (unsigned int*)alloc(64 * 64 * 2);
  pRc0[0] = (unsigned int*)alloc(256 * 64 * 2); pRc0[1] = (unsigned int*)alloc(64 * 64 * 2); pRc0[2] = (unsigned int*)alloc(64 * 64 * 2);
  pRc1[0] = (unsigned int*)alloc(256 * 64 * 2); pRc1[1] = (unsigned int*)alloc(64 * 64 * 2); pRc1[2] = (unsigned int*)alloc(64 * 64 * 2);
  pM[0] = (unsigned int*)alloc(64 * 128 * 2); pM[1] = (unsigned int*)alloc(128 * 128 * 2); pM[2] = (unsigned int*)alloc(128 * 64 * 2);

  float* agg   = (float*)alloc((size_t)N0_NODES * 64 * 4);
  float* x0    = (float*)alloc((size_t)N0_NODES * 64 * 4);
  float* pbuf  = (float*)alloc((size_t)N1_NODES * 64 * 4);
  float* cnt   = (float*)alloc((size_t)N1_NODES * 4);
  float* x1    = (float*)alloc((size_t)N1_NODES * 64 * 4);
  float* xcin  = (float*)alloc((size_t)N1_NODES * 128 * 4);
  float* xc    = (float*)alloc((size_t)N1_NODES * 64 * 4);
  float* xfin  = (float*)alloc((size_t)N0_NODES * 128 * 4);
  float* xf    = (float*)alloc((size_t)N0_NODES * 64 * 4);
  float* stats = (float*)alloc(128 * 4);
  float* ss    = (float*)alloc(128 * 4);

  auto Z = [&](float* p, long n) {
    zero_kernel<<<(unsigned)((n + 255) / 256), 256, 0, stream>>>(p, n);
  };
  auto PACK = [&](const float* W, int K, int N, unsigned int* dst) {
    int total = (K * N) >> 1;
    pack_weight_kernel<<<(total + 255) / 256, 256, 0, stream>>>(W, K, N, dst);
  };
  auto BN = [&](const float* src, float* dst, int nrows, const float* g, const float* b) {
    Z(stats, 128);
    bn_stats_kernel<<<(nrows + 2047) / 2048, 256, 0, stream>>>(src, nrows, stats);
    bn_finalize_kernel<<<1, 64, 0, stream>>>(stats, g, b, (float)nrows, ss);
    long n = (long)nrows * 64;
    bn_apply_kernel<<<(unsigned)((n + 255) / 256), 256, 0, stream>>>(src, ss, dst, n);
  };

  // ---- pack all weights to WMMA fragment layout (f16) ----
  PACK(Lc0W[0], 128, 64, pLc0[0]); PACK(Lc0W[1], 64, 64, pLc0[1]); PACK(Lc0W[2], 64, 64, pLc0[2]);
  PACK(Lc1W[0], 128, 64, pLc1[0]); PACK(Lc1W[1], 64, 64, pLc1[1]); PACK(Lc1W[2], 64, 64, pLc1[2]);
  PACK(Rc0W[0], 256, 64, pRc0[0]); PACK(Rc0W[1], 64, 64, pRc0[1]); PACK(Rc0W[2], 64, 64, pRc0[2]);
  PACK(Rc1W[0], 256, 64, pRc1[0]); PACK(Rc1W[1], 64, 64, pRc1[1]); PACK(Rc1W[2], 64, 64, pRc1[2]);
  PACK(MW[0], 64, 128, pM[0]); PACK(MW[1], 128, 128, pM[1]); PACK(MW[2], 128, 64, pM[2]);

  // ---- down path ----
  // x0 = BN(relu(EdgeConv(X, edges0, Lconv0)))
  Z(agg, (long)N0_NODES * 64);
  edge_conv_kernel<64><<<(E0_EDGES + 63) / 64, 128, 0, stream>>>(
      X, edges0, (long)E0_EDGES, pLc0[0], Lc0b[0], pLc0[1], Lc0b[1], pLc0[2], Lc0b[2], agg);
  BN(agg, x0, N0_NODES, Lnorm0g, Lnorm0b);

  // p = avg_pool(x0, cluster0)
  Z(pbuf, (long)N1_NODES * 64);
  Z(cnt, N1_NODES);
  {
    long n = (long)N0_NODES * 64;
    pool_scatter_kernel<<<(unsigned)((n + 255) / 256), 256, 0, stream>>>(x0, cluster0, pbuf, cnt, N0_NODES);
    long m = (long)N1_NODES * 64;
    pool_div_kernel<<<(unsigned)((m + 255) / 256), 256, 0, stream>>>(pbuf, cnt, m);
  }

  // x1 = BN(relu(EdgeConv(p, edges1, Lconv1)))
  Z(agg, (long)N1_NODES * 64);
  edge_conv_kernel<64><<<(E1_EDGES + 63) / 64, 128, 0, stream>>>(
      pbuf, edges1, (long)E1_EDGES, pLc1[0], Lc1b[0], pLc1[1], Lc1b[1], pLc1[2], Lc1b[2], agg);
  BN(agg, x1, N1_NODES, Lnorm1g, Lnorm1b);

  // ---- up path ----
  // Xc = BN(relu(EdgeConv(cat[x1,x1], edges1, Rconv1)))
  {
    long n = (long)N1_NODES * 128;
    cat_xc_kernel<<<(unsigned)((n + 255) / 256), 256, 0, stream>>>(x1, xcin, n);
  }
  Z(agg, (long)N1_NODES * 64);
  edge_conv_kernel<128><<<(E1_EDGES + 63) / 64, 128, 0, stream>>>(
      xcin, edges1, (long)E1_EDGES, pRc1[0], Rc1b[0], pRc1[1], Rc1b[1], pRc1[2], Rc1b[2], agg);
  BN(agg, xc, N1_NODES, Rnorm1g, Rnorm1b);

  // Xf = BN(relu(EdgeConv(cat[x0, xc[cluster0]], edges0, Rconv0)))
  {
    long n = (long)N0_NODES * 128;
    cat_xf_kernel<<<(unsigned)((n + 255) / 256), 256, 0, stream>>>(x0, xc, cluster0, xfin, n);
  }
  Z(agg, (long)N0_NODES * 64);
  edge_conv_kernel<128><<<(E0_EDGES + 63) / 64, 128, 0, stream>>>(
      xfin, edges0, (long)E0_EDGES, pRc0[0], Rc0b[0], pRc0[1], Rc0b[1], pRc0[2], Rc0b[2], agg);
  BN(agg, xf, N0_NODES, Rnorm0g, Rnorm0b);

  // ---- output MLP ----
  mlp_out_kernel<<<(N0_NODES + 63) / 64, 128, 0, stream>>>(
      xf, N0_NODES, pM[0], Mb[0], pM[1], Mb[1], pM[2], Mb[2], out);
}
